// SAGEModel_16638703305293
// MI455X (gfx1250) — matrix operand order repaired and checked by reading, not compiled, not run on hardware
//
#include <hip/hip_runtime.h>

#define NODES  100000
#define NEDGE  400000
#define DIM    512
#define BN_EPS 1e-5f

typedef __bf16 bf16_t;
typedef __attribute__((ext_vector_type(16))) __bf16 v16bf;
typedef __attribute__((ext_vector_type(8)))  __bf16 v8bf;
typedef __attribute__((ext_vector_type(4)))  __bf16 v4bf;
typedef __attribute__((ext_vector_type(8)))  float  v8f;
typedef __attribute__((ext_vector_type(4)))  float  v4f;

// ---- global-address-space accessors (force global_load/store, not flat) ----
#define GAS __attribute__((address_space(1)))

__device__ __forceinline__ v8bf gload_v8bf(const bf16_t* p) {
    return *(const GAS v8bf*)(unsigned long long)p;
}
__device__ __forceinline__ v4f gload_v4f(const float* p) {
    return *(const GAS v4f*)(unsigned long long)p;
}
__device__ __forceinline__ void gstore_v4f(float* p, v4f v) {
    *(GAS v4f*)(unsigned long long)p = v;
}
__device__ __forceinline__ void gstore_f1(float* p, float v) {
    *(GAS float*)(unsigned long long)p = v;
}
__device__ __forceinline__ void gstore_v4bf(bf16_t* p, v4bf v) {
    *(GAS v4bf*)(unsigned long long)p = v;
}

// ---------------------------------------------------------------- helpers

__global__ void f32_to_bf16_kernel(const float* __restrict__ in,
                                   bf16_t* __restrict__ out, long n) {
    long i = ((long)blockIdx.x * blockDim.x + threadIdx.x) * 4;
    if (i >= n) return;
    v4f v = gload_v4f(in + i);
    v4bf o;
    o[0] = (bf16_t)v[0]; o[1] = (bf16_t)v[1]; o[2] = (bf16_t)v[2]; o[3] = (bf16_t)v[3];
    gstore_v4bf(out + i, o);
}

__global__ void degree_kernel(const int* __restrict__ dst,
                              float* __restrict__ deg) {
    int e = blockIdx.x * blockDim.x + threadIdx.x;
    if (e < NEDGE) atomicAdd(&deg[dst[e]], 1.0f);
}

// one edge spread over 128 threads; each thread moves 4 floats
__global__ void scatter_kernel(const float* __restrict__ x,
                               const int* __restrict__ src,
                               const int* __restrict__ dst,
                               float* __restrict__ agg) {
    int tid = blockIdx.x * blockDim.x + threadIdx.x;
    int e = tid >> 7;
    if (e >= NEDGE) return;
    int d = (tid & 127) << 2;
    long so = (long)src[e] * DIM + d;
    long dofs = (long)dst[e] * DIM + d;
    v4f v = gload_v4f(x + so);
    atomicAdd(agg + dofs + 0, v[0]);
    atomicAdd(agg + dofs + 1, v[1]);
    atomicAdd(agg + dofs + 2, v[2]);
    atomicAdd(agg + dofs + 3, v[3]);
}

// agg / max(deg,1) -> bf16
__global__ void scale_bf16_kernel(const float* __restrict__ agg,
                                  const float* __restrict__ deg,
                                  bf16_t* __restrict__ out) {
    long i = ((long)blockIdx.x * blockDim.x + threadIdx.x) * 4;
    if (i >= (long)NODES * DIM) return;
    int row = (int)(i >> 9);
    float inv = 1.0f / fmaxf(deg[row], 1.0f);
    v4f v = gload_v4f(agg + i);
    v4bf o;
    o[0] = (bf16_t)(v[0] * inv); o[1] = (bf16_t)(v[1] * inv);
    o[2] = (bf16_t)(v[2] * inv); o[3] = (bf16_t)(v[3] * inv);
    gstore_v4bf(out + i, o);
}

// ---------------------------------------------------------------- GEMM
// H = A0 @ B0 + A1 @ B1 + (bl + br), plus column sum/sumsq for BatchNorm.
// A* : [M x 512] bf16 row-major, B* : [512 x 512] bf16 row-major.
// 256 threads = 8 waves (4 along M x 2 along N); wave tile 64x32
// (4 M-frags x 2 N-frags -> 8 WMMA per K-step), block tile 256x64.

__launch_bounds__(256)
__global__ void sage_gemm_kernel(const bf16_t* __restrict__ A0,
                                 const bf16_t* __restrict__ B0,
                                 const bf16_t* __restrict__ A1,
                                 const bf16_t* __restrict__ B1,
                                 const float* __restrict__ bl,
                                 const float* __restrict__ br,
                                 float* __restrict__ H,
                                 float* __restrict__ csum,
                                 float* __restrict__ csq,
                                 int M) {
    const int lane = threadIdx.x & 31;
    const int wave = threadIdx.x >> 5;
    const int wm = wave & 3, wn = wave >> 2;
    const int lm = lane & 15, hf = lane >> 4;

    const int mBase = blockIdx.x * 256 + wm * 64;
    const int n0    = blockIdx.y * 64  + wn * 32;

    int rowA[4];
    #pragma unroll
    for (int mf = 0; mf < 4; ++mf)
        rowA[mf] = min(mBase + mf * 16 + lm, M - 1);

    v8f acc[4][2] = {};

    const bf16_t* Aarr[2] = {A0, A1};
    const bf16_t* Barr[2] = {B0, B1};

    for (int p = 0; p < 2; ++p) {
        const bf16_t* A = Aarr[p];
        const bf16_t* B = Barr[p];
        #pragma unroll 2
        for (int k0 = 0; k0 < DIM; k0 += 32) {
            // B fragment: lane col = lm, K = [k0+8h, +8) ++ [k0+16+8h, +8)
            const bf16_t* brow = B + (size_t)(k0 + 8 * hf) * DIM + n0 + lm;
            v16bf bfr[2];
            #pragma unroll
            for (int nf = 0; nf < 2; ++nf) {
                #pragma unroll
                for (int i = 0; i < 8; ++i) { bfr[nf][i] = brow[(size_t)i * DIM + nf * 16]; bfr[nf][i + 8] = brow[(size_t)(i + 16) * DIM + nf * 16]; }
            }
            // A fragment: lane row = lm, K = [k0+8h, +8) ++ [k0+16+8h, +8)
            #pragma unroll
            for (int mf = 0; mf < 4; ++mf) {
                const bf16_t* arow = A + (size_t)rowA[mf] * DIM + k0 + 8 * hf;
                v8bf lo = gload_v8bf(arow);
                v8bf hi = gload_v8bf(arow + 16);
                v16bf afr;
                #pragma unroll
                for (int i = 0; i < 8; ++i) { afr[i] = lo[i]; afr[i + 8] = hi[i]; }
                #pragma unroll
                for (int nf = 0; nf < 2; ++nf) {
                    acc[mf][nf] = __builtin_amdgcn_wmma_f32_16x16x32_bf16(
                        false, afr, false, bfr[nf], (short)0, acc[mf][nf],
                        false, false);
                }
            }
        }
    }

    // Epilogue: bias, store h, accumulate BatchNorm column statistics.
    // C layout: VGPR r -> row r + 8*hf, col = lm.
    #pragma unroll
    for (int nf = 0; nf < 2; ++nf) {
        const int col = n0 + nf * 16 + lm;
        const float bias = bl[col] + br[col];
        float ps = 0.0f, pq = 0.0f;
        #pragma unroll
        for (int mf = 0; mf < 4; ++mf) {
            const int rb = mBase + mf * 16 + 8 * hf;
            #pragma unroll
            for (int r = 0; r < 8; ++r) {
                const int row = rb + r;
                const float v = acc[mf][nf][r] + bias;
                if (row < M) {
                    gstore_f1(H + (size_t)row * DIM + col, v);
                    ps += v;
                    pq += v * v;
                }
            }
        }
        // lanes L and L^16 hold the same column -> pair-reduce, one atomic/col
        ps += __shfl_xor(ps, 16, 32);
        pq += __shfl_xor(pq, 16, 32);
        if (hf == 0) {
            atomicAdd(&csum[col], ps);
            atomicAdd(&csq[col],  pq);
        }
    }
}

// ---------------------------------------------------------------- BN tail

__global__ void stats_kernel(const float* __restrict__ csum,
                             const float* __restrict__ csq,
                             const float* __restrict__ g,
                             const float* __restrict__ be,
                             float* __restrict__ sc,
                             float* __restrict__ sh) {
    int d = blockIdx.x * blockDim.x + threadIdx.x;
    if (d >= DIM) return;
    const float invN = 1.0f / (float)NODES;
    float mu  = csum[d] * invN;
    float var = fmaxf(csq[d] * invN - mu * mu, 0.0f);
    float s = g[d] * rsqrtf(var + BN_EPS);
    sc[d] = s;
    sh[d] = be[d] - mu * s;
}

// out = x + relu(h*scale + shift); optionally also emit bf16 copy of out
__global__ void bn_relu_res_kernel(const float* __restrict__ x,
                                   const float* __restrict__ h,
                                   const float* __restrict__ sc,
                                   const float* __restrict__ sh,
                                   float* __restrict__ out,
                                   bf16_t* __restrict__ outb) {
    long i = ((long)blockIdx.x * blockDim.x + threadIdx.x) * 4;
    if (i >= (long)NODES * DIM) return;
    int col = (int)(i & (DIM - 1));
    v4f hv = gload_v4f(h + i);
    v4f xv = gload_v4f(x + i);
    v4f o;
    o[0] = xv[0] + fmaxf(0.0f, hv[0] * sc[col + 0] + sh[col + 0]);
    o[1] = xv[1] + fmaxf(0.0f, hv[1] * sc[col + 1] + sh[col + 1]);
    o[2] = xv[2] + fmaxf(0.0f, hv[2] * sc[col + 2] + sh[col + 2]);
    o[3] = xv[3] + fmaxf(0.0f, hv[3] * sc[col + 3] + sh[col + 3]);
    gstore_v4f(out + i, o);
    if (outb) {
        v4bf b;
        b[0] = (bf16_t)o[0]; b[1] = (bf16_t)o[1]; b[2] = (bf16_t)o[2]; b[3] = (bf16_t)o[3];
        gstore_v4bf(outb + i, b);
    }
}

// ---------------------------------------------------------------- launch

extern "C" void kernel_launch(void* const* d_in, const int* in_sizes, int n_in,
                              void* d_out, int out_size, void* d_ws, size_t ws_size,
                              hipStream_t stream) {
    (void)in_sizes; (void)n_in; (void)out_size; (void)ws_size;

    const int*   ei  = (const int*)d_in[0];
    const int*   src = ei;
    const int*   dst = ei + NEDGE;
    const float* emb = (const float*)d_in[1];
    const float *Wl0 = (const float*)d_in[2],  *bl0 = (const float*)d_in[3];
    const float *Wr0 = (const float*)d_in[4],  *br0 = (const float*)d_in[5];
    const float *g0  = (const float*)d_in[6],  *be0 = (const float*)d_in[7];
    const float *Wl1 = (const float*)d_in[8],  *bl1 = (const float*)d_in[9];
    const float *Wr1 = (const float*)d_in[10], *br1 = (const float*)d_in[11];
    const float *g1  = (const float*)d_in[12], *be1 = (const float*)d_in[13];

    const size_t SZF = (size_t)NODES * DIM * sizeof(float);   // 204.8 MB
    const size_t SZB = (size_t)NODES * DIM * sizeof(bf16_t);  // 102.4 MB
    const size_t SZW = (size_t)DIM * DIM * sizeof(bf16_t);    // 512 KB

    char* w = (char*)d_ws;
    float*  aggH = (float*)(w);                      // agg fp32, reused as h
    bf16_t* aggB = (bf16_t*)(w + SZF);               // bf16 aggregated feats
    bf16_t* xB   = (bf16_t*)(w + SZF + SZB);         // bf16 root feats
    float*  x1   = (float*)(w + SZF + 2 * SZB);      // layer-0 output fp32
    char*   p    = w + 2 * SZF + 2 * SZB;
    float*  deg  = (float*)p;  p += 400384;          // N floats (padded)
    float*  csum = (float*)p;                        // 4 * D floats
    float*  csq  = csum + DIM;
    float*  bsc  = csq + DIM;
    float*  bsh  = bsc + DIM;
    p += 4 * DIM * sizeof(float);
    bf16_t* Wl0b = (bf16_t*)p; p += SZW;
    bf16_t* Wr0b = (bf16_t*)p; p += SZW;
    bf16_t* Wl1b = (bf16_t*)p; p += SZW;
    bf16_t* Wr1b = (bf16_t*)p; p += SZW;

    const long ND = (long)NODES * DIM;
    const int ND4B = (int)(ND / 4 / 256);                    // 50000
    const int DD4B = (int)((DIM * DIM / 4 + 255) / 256);     // 256
    const dim3 gemmGrid((NODES + 255) / 256, DIM / 64);      // 391 x 8

    // precision conversion (weights + root features)
    f32_to_bf16_kernel<<<ND4B, 256, 0, stream>>>(emb, xB, ND);
    f32_to_bf16_kernel<<<DD4B, 256, 0, stream>>>(Wl0, Wl0b, (long)DIM * DIM);
    f32_to_bf16_kernel<<<DD4B, 256, 0, stream>>>(Wr0, Wr0b, (long)DIM * DIM);
    f32_to_bf16_kernel<<<DD4B, 256, 0, stream>>>(Wl1, Wl1b, (long)DIM * DIM);
    f32_to_bf16_kernel<<<DD4B, 256, 0, stream>>>(Wr1, Wr1b, (long)DIM * DIM);

    // degrees (shared by both layers)
    (void)hipMemsetAsync(deg, 0, NODES * sizeof(float), stream);
    degree_kernel<<<(NEDGE + 255) / 256, 256, 0, stream>>>(dst, deg);

    // ---------------- layer 0 ----------------
    (void)hipMemsetAsync(aggH, 0, SZF, stream);
    scatter_kernel<<<NEDGE * 128 / 256, 256, 0, stream>>>(emb, src, dst, aggH);
    scale_bf16_kernel<<<ND4B, 256, 0, stream>>>(aggH, deg, aggB);
    (void)hipMemsetAsync(csum, 0, 2 * DIM * sizeof(float), stream);
    sage_gemm_kernel<<<gemmGrid, 256, 0, stream>>>(aggB, Wl0b, xB, Wr0b,
                                                   bl0, br0, aggH, csum, csq, NODES);
    stats_kernel<<<2, 256, 0, stream>>>(csum, csq, g0, be0, bsc, bsh);
    bn_relu_res_kernel<<<ND4B, 256, 0, stream>>>(emb, aggH, bsc, bsh, x1, xB);

    // ---------------- layer 1 ----------------
    (void)hipMemsetAsync(aggH, 0, SZF, stream);
    scatter_kernel<<<NEDGE * 128 / 256, 256, 0, stream>>>(x1, src, dst, aggH);
    scale_bf16_kernel<<<ND4B, 256, 0, stream>>>(aggH, deg, aggB);
    (void)hipMemsetAsync(csum, 0, 2 * DIM * sizeof(float), stream);
    sage_gemm_kernel<<<gemmGrid, 256, 0, stream>>>(aggB, Wl1b, xB, Wr1b,
                                                   bl1, br1, aggH, csum, csq, NODES);
    stats_kernel<<<2, 256, 0, stream>>>(csum, csq, g1, be1, bsc, bsh);
    bn_relu_res_kernel<<<ND4B, 256, 0, stream>>>(x1, aggH, bsc, bsh,
                                                 (float*)d_out, nullptr);
}
